// FastLLaDAMoE_4647154615602
// MI455X (gfx1250) — compile-verified, matrix operand order checked by reading, not run on hardware
//
#include <hip/hip_runtime.h>
#include <stdint.h>
#include <stddef.h>

// ---------------------------------------------------------------------------
// MoE (LLaDA-style) forward for MI455X / gfx1250.
//
// Strategy (see analysis): route -> compact -> per-expert tiled SwiGLU FFN
// using V_WMMA_F32_16X16X32_BF16 with a bf16 hi/lo error-compensated split
// (acc += Ahi*Bhi + Ahi*Blo + Alo*Bhi) for ~fp32 accuracy at ~3x bf16 cost,
// which is ~3x faster than the native f32 WMMA path (K=4) on CDNA5.
// Weights/activations are pre-split once per call (bandwidth bound).
// ---------------------------------------------------------------------------

#define NUM_EXPERTS 8
#define TOPK        2
#define HID         2048
#define INTER       2816
#define INTER2      5632              // 2*INTER
#define NTOK        8192              // BSZ*SEQ
#define LDSTRIDE    2824              // INTER + 8 bf16 pad -> bank-conflict-free

typedef __attribute__((ext_vector_type(16))) __bf16 v16bf;
typedef __attribute__((ext_vector_type(8)))  float  v8f;

// ---------------- bf16 helpers (RNE) ----------------
__device__ __forceinline__ unsigned short f32_to_bf16(float f) {
    unsigned int u = __float_as_uint(f);
    u += 0x7fffu + ((u >> 16) & 1u);
    return (unsigned short)(u >> 16);
}
__device__ __forceinline__ float bf16_to_f32(unsigned short h) {
    return __uint_as_float(((unsigned int)h) << 16);
}

// Load one 16x(16|32) WMMA operand fragment: per-lane 8 bf16 at +0 and 8 at +16
// (matches the CDNA5 ISA 16-bit A/B layout with the (lane>>4)*8 K-split applied
// by the caller through the element offset).
union FragU { v16bf v; uint4 q[2]; };
__device__ __forceinline__ v16bf load_frag(const unsigned short* p) {
    FragU u;
    u.q[0] = *(const uint4*)(p);
    u.q[1] = *(const uint4*)(p + 16);
    return u.v;
}

__device__ __forceinline__ v8f wmma_bf16(v16bf a, v16bf b, v8f c) {
    // emits v_wmma_f32_16x16x32_bf16
    return __builtin_amdgcn_wmma_f32_16x16x32_bf16(false, a, false, b,
                                                   (short)0, c, false, false);
}

// ---------------- fp32 -> bf16 hi/lo split ----------------
__global__ void split_bf16_kernel(const float* __restrict__ src,
                                  unsigned short* __restrict__ hi,
                                  unsigned short* __restrict__ lo,
                                  size_t n) {
    size_t i = (size_t)blockIdx.x * blockDim.x + threadIdx.x;
    size_t stride = (size_t)gridDim.x * blockDim.x;
    for (; i < n; i += stride) {
        float v = src[i];
        unsigned short h = f32_to_bf16(v);
        hi[i] = h;
        lo[i] = f32_to_bf16(v - bf16_to_f32(h));
    }
}

// ---------------- router: softmax + bias + top2 -> dense comb[T,8] ----------
__global__ void router_kernel(const float* __restrict__ x,
                              const float* __restrict__ gw,
                              const float* __restrict__ gb,
                              const float* __restrict__ ebias,
                              float* __restrict__ comb) {
    const int wave = threadIdx.x >> 5;
    const int lane = threadIdx.x & 31;
    const int t = blockIdx.x * 8 + wave;       // one wave32 per token
    const float* xr = x + (size_t)t * HID;

    float acc[NUM_EXPERTS];
#pragma unroll
    for (int e = 0; e < NUM_EXPERTS; ++e) acc[e] = 0.f;

    for (int k = lane; k < HID; k += 32) {
        float xv = xr[k];
#pragma unroll
        for (int e = 0; e < NUM_EXPERTS; ++e)
            acc[e] += xv * gw[e * HID + k];
    }
#pragma unroll
    for (int e = 0; e < NUM_EXPERTS; ++e)
        for (int off = 16; off; off >>= 1)
            acc[e] += __shfl_xor(acc[e], off, 32);

    // every lane now holds all 8 logits; fp32 softmax + bias + top-2
    float mx = -3.4e38f;
#pragma unroll
    for (int e = 0; e < NUM_EXPERTS; ++e) { acc[e] += gb[e]; mx = fmaxf(mx, acc[e]); }
    float p[NUM_EXPERTS], sum = 0.f;
#pragma unroll
    for (int e = 0; e < NUM_EXPERTS; ++e) { p[e] = __expf(acc[e] - mx); sum += p[e]; }
    float inv = 1.f / sum;
#pragma unroll
    for (int e = 0; e < NUM_EXPERTS; ++e) p[e] = p[e] * inv + ebias[e];

    int i1 = 0; float b1 = p[0];
#pragma unroll
    for (int e = 1; e < NUM_EXPERTS; ++e) if (p[e] > b1) { b1 = p[e]; i1 = e; }
    int i2 = -1; float b2 = -3.4e38f;
#pragma unroll
    for (int e = 0; e < NUM_EXPERTS; ++e)
        if (e != i1 && p[e] > b2) { b2 = p[e]; i2 = e; }

    if (lane < NUM_EXPERTS) {
        float w = (lane == i1) ? b1 : ((lane == i2) ? b2 : 0.f);
        comb[(size_t)t * NUM_EXPERTS + lane] = w;
    }
}

// ---------------- deterministic compaction: one wave per expert -------------
__global__ void scan_kernel(const float* __restrict__ comb,
                            int* __restrict__ lists,
                            int* __restrict__ counts) {
    const int e = threadIdx.x >> 5;            // 8 waves = 8 experts
    const int lane = threadIdx.x & 31;
    int cnt = 0;
    for (int t0 = 0; t0 < NTOK; t0 += 32) {
        int t = t0 + lane;
        bool f = comb[(size_t)t * NUM_EXPERTS + e] != 0.f;
        unsigned mask = (unsigned)__ballot(f);                  // wave32
        int pos = __popc(mask & ((1u << lane) - 1u));
        if (f) lists[e * NTOK + cnt + pos] = t;
        cnt += __popc(mask);
    }
    if (lane == 0) counts[e] = cnt;
}

// ---------------- per-expert tiled SwiGLU FFN (WMMA bf16x3) -----------------
__global__ void __launch_bounds__(256)
moe_ffn_kernel(const unsigned short* __restrict__ x_hi,
               const unsigned short* __restrict__ x_lo,
               const unsigned short* __restrict__ wgu_hi,
               const unsigned short* __restrict__ wgu_lo,
               const unsigned short* __restrict__ wd_hi,
               const unsigned short* __restrict__ wd_lo,
               const float* __restrict__ comb,
               const int* __restrict__ lists,
               const int* __restrict__ counts,
               float* __restrict__ out) {
    const int e = blockIdx.y;
    const int cnt = counts[e];
    const int base = blockIdx.x * 16;
    if (base >= cnt) return;                   // uniform early exit

    __shared__ int   tokens_s[16];
    __shared__ float scales_s[16];
    extern __shared__ __align__(16) unsigned short inter_s[]; // [2][16][LDSTRIDE]
    unsigned short* inter_hi = inter_s;
    unsigned short* inter_lo = inter_s + 16 * LDSTRIDE;

    if (threadIdx.x < 16) {
        int idx = base + threadIdx.x;
        int src = (idx < cnt) ? idx : (cnt - 1);
        int tok = lists[e * NTOK + src];
        tokens_s[threadIdx.x] = tok;
        scales_s[threadIdx.x] = (idx < cnt) ? comb[(size_t)tok * NUM_EXPERTS + e] : 0.f;
    }
    __syncthreads();

    const int wave  = threadIdx.x >> 5;        // 8 waves per block
    const int lane  = threadIdx.x & 31;
    const int nlane = lane & 15;               // row (A) / column (B) index
    const int koff  = (lane >> 4) << 3;        // 0 or 8: ISA K-split per lane half

    const size_t arow = (size_t)tokens_s[nlane] * HID;
    const unsigned short* Ahi = x_hi + arow;
    const unsigned short* Alo = x_lo + arow;

    // ---- GEMM1: [16,HID] x Wgu^T -> gate/up, SwiGLU, split into LDS ----
    for (int nc = wave; nc < INTER / 16; nc += 8) {
        const int ng = nc * 16;
        const unsigned short* Bgh = wgu_hi + ((size_t)e * INTER2 + ng + nlane) * HID;
        const unsigned short* Bgl = wgu_lo + ((size_t)e * INTER2 + ng + nlane) * HID;
        const unsigned short* Buh = wgu_hi + ((size_t)e * INTER2 + INTER + ng + nlane) * HID;
        const unsigned short* Bul = wgu_lo + ((size_t)e * INTER2 + INTER + ng + nlane) * HID;
        v8f cg = {}; v8f cu = {};
        for (int k0 = 0; k0 < HID; k0 += 32) {
            const int o = k0 + koff;
            v16bf ah  = load_frag(Ahi + o);
            v16bf al  = load_frag(Alo + o);
            v16bf bgh = load_frag(Bgh + o);
            v16bf bgl = load_frag(Bgl + o);
            v16bf buh = load_frag(Buh + o);
            v16bf bul = load_frag(Bul + o);
            cg = wmma_bf16(ah, bgh, cg);  cu = wmma_bf16(ah, buh, cu);
            cg = wmma_bf16(ah, bgl, cg);  cu = wmma_bf16(ah, bul, cu);
            cg = wmma_bf16(al, bgh, cg);  cu = wmma_bf16(al, buh, cu);
        }
#pragma unroll
        for (int r = 0; r < 8; ++r) {
            // C/D layout: element (r,lane) -> M = r + 8*(lane>=16), N = lane&15
            float g = cg[r], u = cu[r];
            float s = (g / (1.f + __expf(-g))) * u;        // silu(g)*u
            unsigned short h = f32_to_bf16(s);
            unsigned short l = f32_to_bf16(s - bf16_to_f32(h));
            int mr  = r + koff;                             // 0..15
            int col = ng + nlane;
            inter_hi[mr * LDSTRIDE + col] = h;
            inter_lo[mr * LDSTRIDE + col] = l;
        }
    }
    __syncthreads();

    // ---- GEMM2: inter[16,INTER] x Wd^T -> out[16,HID], scaled atomic add ----
    const unsigned short* Aih = inter_hi + nlane * LDSTRIDE;
    const unsigned short* Ail = inter_lo + nlane * LDSTRIDE;
    for (int nc = wave; nc < HID / 16; nc += 8) {
        const int nh = nc * 16;
        const unsigned short* Bh = wd_hi + ((size_t)e * HID + nh + nlane) * INTER;
        const unsigned short* Bl = wd_lo + ((size_t)e * HID + nh + nlane) * INTER;
        v8f c = {};
        for (int i0 = 0; i0 < INTER; i0 += 32) {
            const int o = i0 + koff;
            v16bf ah = load_frag(Aih + o);      // ds_load_b128 from LDS
            v16bf al = load_frag(Ail + o);
            v16bf bh = load_frag(Bh + o);
            v16bf bl = load_frag(Bl + o);
            c = wmma_bf16(ah, bh, c);
            c = wmma_bf16(ah, bl, c);
            c = wmma_bf16(al, bh, c);
        }
#pragma unroll
        for (int r = 0; r < 8; ++r) {
            int mr = r + koff;
            float v = c[r] * scales_s[mr];
            atomicAdd(&out[(size_t)tokens_s[mr] * HID + nh + nlane], v);
        }
    }
}

// ---------------------------------------------------------------------------
extern "C" void kernel_launch(void* const* d_in, const int* in_sizes, int n_in,
                              void* d_out, int out_size, void* d_ws, size_t ws_size,
                              hipStream_t stream) {
    (void)in_sizes; (void)n_in; (void)ws_size;
    const float* x      = (const float*)d_in[0];   // [4,2048,2048]
    const float* gate_w = (const float*)d_in[1];   // [8,2048]
    const float* gate_b = (const float*)d_in[2];   // [8]
    const float* ebias  = (const float*)d_in[3];   // [8]
    const float* wgu    = (const float*)d_in[4];   // [8,5632,2048]
    const float* wd     = (const float*)d_in[5];   // [8,2048,2816]
    float* out = (float*)d_out;

    // workspace carve-up (all 256B aligned)
    char* ws = (char*)d_ws;
    const size_t n_wgu = (size_t)NUM_EXPERTS * INTER2 * HID;   // 92,274,688
    const size_t n_wd  = (size_t)NUM_EXPERTS * HID * INTER;    // 46,137,344
    const size_t n_x   = (size_t)NTOK * HID;                   // 16,777,216
    size_t off = 0;
    unsigned short* wgu_hi = (unsigned short*)(ws + off); off += n_wgu * 2;
    unsigned short* wgu_lo = (unsigned short*)(ws + off); off += n_wgu * 2;
    unsigned short* wd_hi  = (unsigned short*)(ws + off); off += n_wd * 2;
    unsigned short* wd_lo  = (unsigned short*)(ws + off); off += n_wd * 2;
    unsigned short* x_hi   = (unsigned short*)(ws + off); off += n_x * 2;
    unsigned short* x_lo   = (unsigned short*)(ws + off); off += n_x * 2;
    float*          combp  = (float*)(ws + off);          off += (size_t)NTOK * NUM_EXPERTS * 4;
    int*            listsp = (int*)(ws + off);            off += (size_t)NUM_EXPERTS * NTOK * 4;
    int*            cntp   = (int*)(ws + off);            off += 256;

    hipMemsetAsync(d_out, 0, (size_t)out_size * sizeof(float), stream);

    split_bf16_kernel<<<8192, 256, 0, stream>>>(wgu, wgu_hi, wgu_lo, n_wgu);
    split_bf16_kernel<<<8192, 256, 0, stream>>>(wd,  wd_hi,  wd_lo,  n_wd);
    split_bf16_kernel<<<4096, 256, 0, stream>>>(x,   x_hi,   x_lo,   n_x);

    router_kernel<<<NTOK / 8, 256, 0, stream>>>(x, gate_w, gate_b, ebias, combp);
    scan_kernel<<<1, 256, 0, stream>>>(combp, listsp, cntp);

    dim3 grid(NTOK / 16, NUM_EXPERTS);   // worst case: all tokens to one expert
    const size_t lds_bytes = (size_t)2 * 16 * LDSTRIDE * sizeof(unsigned short); // 180,736
    moe_ffn_kernel<<<grid, 256, lds_bytes, stream>>>(x_hi, x_lo, wgu_hi, wgu_lo,
                                                     wd_hi, wd_lo, combp, listsp,
                                                     cntp, out);
}
